// DisGraphAF_20925080666411
// MI455X (gfx1250) — matrix operand order repaired and checked by reading, not compiled, not run on hardware
//
#include <hip/hip_runtime.h>
#include <hip/hip_bf16.h>
#include <cstdint>

// ---------------- problem constants ----------------
#define B_    16
#define N_    38
#define NT_   9
#define ET_   4
#define NE_   703        // N*(N-1)/2
#define R_    741        // N + NE
#define H_    128
#define L_    12
#define EDIM_ 3

typedef __attribute__((ext_vector_type(16))) _Float16 v16h;
typedef __attribute__((ext_vector_type(8)))  float    v8f;
typedef __attribute__((ext_vector_type(4)))  unsigned int u32x4;
typedef __attribute__((ext_vector_type(8)))  int      i32x8;
typedef __attribute__((ext_vector_type(4)))  int      i32x4;

// ---------------- WMMA helpers (CDNA5 wave32, 16x16x32 f16 -> f32) -------
__device__ __forceinline__ v8f wmma_f16(v16h a, v16h b, v8f c) {
  return __builtin_amdgcn_wmma_f32_16x16x32_f16(
      /*neg_a=*/false, a, /*neg_b=*/false, b,
      /*c_mod=*/(short)0, c, /*reuse_a=*/false, /*reuse_b=*/false);
}

// A fragment (16x32 f16), row-major source base[row*ld + k].
// ISA layout: lane L -> row M0 + L%16 ; half t -> K = K0 + (t<8?0:16) + 8*(L/16) + t%8
__device__ __forceinline__ v16h frag_a_ld(const _Float16* __restrict__ base,
                                          int ld, int M0, int K0, int lane) {
  const int g = lane >> 4;
  const int m = M0 + (lane & 15);
  const _Float16* p = base + m * ld + K0 + 8 * g;
  v16h a;
#pragma unroll
  for (int i = 0; i < 8; ++i) { a[i] = p[i]; a[8 + i] = p[16 + i]; }
  return a;
}

// B fragment (32x16 f16); element (k,n) stored at base[n*ld + k] (k contiguous).
// ISA layout: lane L -> col N0 + L%16 ; half t -> K = K0 + (t<8?0:16) + 8*(L/16) + t%8
__device__ __forceinline__ v16h frag_b_ld(const _Float16* __restrict__ base,
                                          int ld, int K0, int N0, int lane) {
  const int g = lane >> 4;
  const int n = N0 + (lane & 15);
  const _Float16* p = base + n * ld + K0 + 8 * g;
  v16h b;
#pragma unroll
  for (int t = 0; t < 8; ++t) { b[t] = p[t]; b[8 + t] = p[16 + t]; }
  return b;
}

// C/D layout: lane L -> col N0 + L%16 ; VGPR v -> row M0 + 8*(L/16) + v
__device__ __forceinline__ void store_c_f16_relu(_Float16* dst, int ld, int M0,
                                                 int N0, int lane, v8f c,
                                                 int maxrow) {
  const int g = lane >> 4;
  const int n = N0 + (lane & 15);
#pragma unroll
  for (int v = 0; v < 8; ++v) {
    int row = M0 + 8 * g + v;
    float val = fmaxf(c[v], 0.0f);
    dst[row * ld + n] = (row < maxrow) ? (_Float16)val : (_Float16)0.0f;
  }
}

__device__ __forceinline__ void store_c_f16_T(_Float16* dst, int ldk, int M0,
                                              int N0, int lane, v8f c) {
  const int g = lane >> 4;
  const int n = N0 + (lane & 15);
#pragma unroll
  for (int v = 0; v < 8; ++v) {
    int row = M0 + 8 * g + v;
    dst[n * ldk + row] = (_Float16)c[v];
  }
}

// ---------------- TDM: 1-D contiguous copy of n16 f16 elements to LDS -----
// Builds a Tensor DMA Descriptor per CDNA5 ISA:
//  group0: count=1 | lds_addr | global_addr(57b) | type=2
//  group1: data_size=1(2B) | tensor_dim0=n16 | tile_dim0=n16 (1-D tile)
__device__ __forceinline__ void tdm_load_to_lds(unsigned lds_off,
                                                const void* gsrc, int n16) {
  unsigned long long ga = (unsigned long long)gsrc;
  u32x4 g0;
  g0[0] = 1u;                                        // count=1, user mode
  g0[1] = lds_off;                                   // lds byte address
  g0[2] = (unsigned)(ga & 0xffffffffu);              // global_addr[31:0]
  g0[3] = (unsigned)((ga >> 32) & 0x01ffffffu) | (2u << 30); // [56:32] | type=2
  i32x8 g1;
  g1[0] = (int)(1u << 16);                           // data_size = 1 -> 2 bytes
  g1[1] = (int)((unsigned)(n16 & 0xffff) << 16);     // tensor_dim0[15:0]
  g1[2] = (int)(((unsigned)n16 >> 16) & 0xffffu);    // tensor_dim0[31:16]
  g1[3] = (int)((unsigned)(n16 & 0xffff) << 16);     // tile_dim0[15:0] (<=65535)
  g1[4] = 0;                                         // tile_dim1/2 = 0 (1-D)
  g1[5] = n16;                                       // tensor_dim0_stride[31:0]
  g1[6] = 0;
  g1[7] = 0;
  i32x4 z4 = {0, 0, 0, 0};
#if defined(__clang_major__) && (__clang_major__ >= 23)
  i32x8 z8 = {0, 0, 0, 0, 0, 0, 0, 0};
  __builtin_amdgcn_tensor_load_to_lds(g0, g1, z4, z4, z8, 0);
#else
  __builtin_amdgcn_tensor_load_to_lds(g0, g1, z4, z4, 0);
#endif
}

// ---------------- kernel 0: weight f16 conversion + BN accumulator zero ----
__global__ void k_prep(const float* __restrict__ gc2, const float* __restrict__ gc3,
                       const float* __restrict__ ew1, const float* __restrict__ nw1,
                       _Float16* gc2_16, _Float16* gc3_16, _Float16* ew1_16,
                       _Float16* nw1_16, float* bn_zero /*256 floats*/) {
  const int stride = gridDim.x * blockDim.x;
  const int total = 49152 + 49152 + 589824 + 196608 + 256;
  for (int t = blockIdx.x * blockDim.x + threadIdx.x; t < total; t += stride) {
    int u = t;
    if (u < 49152)  { gc2_16[u] = (_Float16)gc2[u]; continue; } u -= 49152;
    if (u < 49152)  { gc3_16[u] = (_Float16)gc3[u]; continue; } u -= 49152;
    if (u < 589824) { ew1_16[u] = (_Float16)ew1[u]; continue; } u -= 589824;
    if (u < 196608) { nw1_16[u] = (_Float16)nw1[u]; continue; } u -= 196608;
    bn_zero[u] = 0.0f;
  }
}

// ---------------- kernel 1: fused masked RGCN per graph --------------------
// one workgroup (256 threads / 8 waves) per (b, r) graph.
// LDS (dynamic): sh_a 3*48*64 f16 | sh_sT 384*64 f16 | sh_h 48*128 f16 | sh_h0 38*12 f32
// sh_z (48*128 f32) aliases sh_sT after the final einsum.
__global__ __launch_bounds__(256) void k_rgcn(
    const float* __restrict__ x, const float* __restrict__ adj,
    const unsigned char* __restrict__ mask_node,
    const unsigned char* __restrict__ mask_edge, const int* __restrict__ isel,
    const float* __restrict__ emb_w, const float* __restrict__ gc1_w,
    const _Float16* __restrict__ gc2_16, const _Float16* __restrict__ gc3_16,
    float* bn_sum, float* bn_sumsq, float* graph_sum, float* selA, float* selB) {
  extern __shared__ char smem[];
  _Float16* sh_a  = (_Float16*)(smem);                        // 18432 B
  _Float16* sh_sT = (_Float16*)(smem + 18432);                // 49152 B
  _Float16* sh_h  = (_Float16*)(smem + 18432 + 49152);        // 12288 B
  float*    sh_h0 = (float*)   (smem + 18432 + 49152 + 12288);//  1824 B
  float*    sh_z  = (float*)   (smem + 18432);                // alias sh_sT

  const int g = blockIdx.x;
  const int b = g / R_;
  const int r = g % R_;
  const int tid = threadIdx.x, wave = tid >> 5, lane = tid & 31;

  // ---- phase A: masked inputs, embed, gc1 s-matrix (K=9, VALU) ----
  for (int t = tid; t < 3 * 48 * 64; t += 256) {
    int e = t / (48 * 64), rem = t % (48 * 64);
    int i = rem / 64, j = rem % 64;
    float v = 0.0f;
    if (i < N_ && j < N_ && mask_edge[r * N_ * N_ + i * N_ + j])
      v = adj[((b * ET_ + e) * N_ + i) * N_ + j];
    sh_a[t] = (_Float16)v;
  }
  for (int t = tid; t < 384 * 26; t += 256) {          // zero K-pad rows 38..63
    int col = t / 26, j = 38 + t % 26;
    sh_sT[col * 64 + j] = (_Float16)0.0f;
  }
  for (int t = tid; t < N_ * NT_; t += 256) {          // h0 = xr @ emb_w^T
    int i = t / NT_, tt = t % NT_;
    float acc = 0.0f;
    if (mask_node[r * N_ + i]) {
#pragma unroll
      for (int u = 0; u < NT_; ++u)
        acc += x[(b * N_ + i) * NT_ + u] * emb_w[tt * NT_ + u];
    }
    sh_h0[i * 12 + tt] = acc;
  }
  __syncthreads();
  for (int t = tid; t < N_ * 384; t += 256) {          // s1 = h0 @ gc1_w^T
    int j = t / 384, m = t % 384;
    float acc = 0.0f;
#pragma unroll
    for (int u = 0; u < NT_; ++u)
      acc += sh_h0[j * 12 + u] * gc1_w[m * NT_ + u];
    sh_sT[m * 64 + j] = (_Float16)acc;
  }
  __syncthreads();

  // ---- 3x (einsum) interleaved with 2x (GEMM s = h @ W^T), all WMMA ----
  v8f zacc[3] = {v8f{}, v8f{}, v8f{}};
  for (int l = 0; l < 3; ++l) {
    // einsum: z[i,o] = sum_e sum_j a[e,i,j] * s[j, e*128+o]
    for (int tt0 = 0; tt0 < 3; ++tt0) {
      int tt = wave * 3 + tt0;
      int Mt = tt >> 3, Nt = tt & 7;
      v8f acc = {};
#pragma unroll
      for (int e = 0; e < 3; ++e)
#pragma unroll
        for (int kk = 0; kk < 2; ++kk) {
          v16h af = frag_a_ld(sh_a + e * 48 * 64, 64, Mt * 16, kk * 32, lane);
          v16h bf = frag_b_ld(sh_sT, 64, kk * 32, e * 128 + Nt * 16, lane);
          acc = wmma_f16(af, bf, acc);
        }
      if (l < 2) store_c_f16_relu(sh_h, 128, Mt * 16, Nt * 16, lane, acc, N_);
      else       zacc[tt0] = acc;
    }
    __syncthreads();
    if (l == 2) break;
    // GEMM: s = h @ W^T  (W f16, out-channel-major [384][128], k contiguous)
    const _Float16* W = (l == 0) ? gc2_16 : gc3_16;
    for (int tt0 = 0; tt0 < 9; ++tt0) {
      int tt = wave * 9 + tt0;
      int Mt = tt / 24, Nt = tt % 24;
      v8f acc = {};
#pragma unroll
      for (int ks = 0; ks < 4; ++ks) {
        v16h af = frag_a_ld(sh_h, 128, Mt * 16, ks * 32, lane);
        v16h bf = frag_b_ld(W, 128, ks * 32, Nt * 16, lane);
        acc = wmma_f16(af, bf, acc);
      }
      store_c_f16_T(sh_sT, 64, Mt * 16, Nt * 16, lane, acc);
    }
    __syncthreads();
  }

  // spill final z (f32) into LDS aliasing sh_sT (all reads of sh_sT are done)
  for (int tt0 = 0; tt0 < 3; ++tt0) {
    int tt = wave * 3 + tt0;
    int Mt = tt >> 3, Nt = tt & 7;
    int gg = lane >> 4, n = Nt * 16 + (lane & 15);
#pragma unroll
    for (int v = 0; v < 8; ++v) {
      int row = Mt * 16 + 8 * gg + v;
      if (row < N_) sh_z[row * 128 + n] = zacc[tt0][v];
    }
  }
  __syncthreads();

  // ---- reductions: BN partial sums + per-graph summaries ----
  if (tid < 128) {
    int d = tid;
    float s1 = 0.0f, s2 = 0.0f;
    for (int i = 0; i < N_; ++i) {
      float zv = sh_z[i * 128 + d];
      s1 += zv; s2 += zv * zv;
    }
    atomicAdd(&bn_sum[d], s1);
    atomicAdd(&bn_sumsq[d], s2);
    graph_sum[g * 128 + d] = s1;
    if (r >= N_) {
      int e = r - N_, eg = b * NE_ + e;
      int iu = isel[2 * e], ju = isel[2 * e + 1];
      selA[eg * 128 + d] = sh_z[iu * 128 + d];
      selB[eg * 128 + d] = sh_z[ju * 128 + d];
    }
  }
}

// ---------------- kernel 2: finalize BN scale/shift ------------------------
__global__ void k_bn(const float* bn_sum, const float* bn_sumsq,
                     const float* gamma, const float* beta,
                     float* scale, float* shift) {
  int d = threadIdx.x;
  if (d < 128) {
    const float cnt = (float)(B_ * R_ * N_);
    float m  = bn_sum[d] / cnt;
    float vr = bn_sumsq[d] / cnt - m * m;
    float sc = gamma[d] * rsqrtf(vr + 1e-5f);
    scale[d] = sc;
    shift[d] = beta[d] - m * sc;
  }
}

// ---------------- kernel 3: node flow (12 layers -> shift of x_deq) --------
// Per-layer weight block (128x128 f16 = 32KB) is DMA'd into LDS by the
// Tensor Data Mover (wave 0 issues, TENSORcnt-waited), WMMA consumes from LDS.
// Dynamic LDS layout: sx 12288 | sh_w 32768 | sh 24576 | slg 2304 | sks 192
__global__ __launch_bounds__(256) void k_node_flow(
    const float* __restrict__ graph_sum, const float* __restrict__ scale,
    const float* __restrict__ shift, const _Float16* __restrict__ nw1_16,
    const float* __restrict__ node_b1, const float* __restrict__ node_w2,
    const float* __restrict__ node_b2, const float* __restrict__ x_deq,
    float* __restrict__ out) {
  extern __shared__ char smem[];
  _Float16* sx   = (_Float16*)(smem);             // 48*128 f16
  _Float16* sh_w = (_Float16*)(smem + 12288);     // 128*128 f16 (layer weights)
  float*    sh   = (float*)   (smem + 45056);     // 48*128 f32
  float*    slg  = (float*)   (smem + 69632);     // 48*12 f32
  int*      sks  = (int*)     (smem + 71936);     // 48 int
  const int tid = threadIdx.x, wave = tid >> 5, lane = tid & 31;
  const int base = blockIdx.x * 48;

  for (int t = tid; t < 48 * 128; t += 256) {
    int row = t >> 7, d = t & 127;
    int gr = base + row;
    float v = 0.0f;
    if (gr < B_ * N_) {
      int bb = gr / N_, n = gr % N_;
      v = graph_sum[(bb * R_ + n) * 128 + d] * scale[d] + (float)N_ * shift[d];
    }
    sx[t] = (_Float16)v;
  }
  if (tid < 48) sks[tid] = 0;
  __syncthreads();

  for (int i = 0; i < L_; ++i) {
    // TDM: stage this layer's 128x128 f16 weight block into LDS
    if (wave == 0) {
      unsigned lds_off = (unsigned)(uintptr_t)sh_w;
      tdm_load_to_lds(lds_off, nw1_16 + i * 128 * 128, 128 * 128);
      __builtin_amdgcn_s_wait_tensorcnt(0);
    }
    __syncthreads();
    for (int tt0 = 0; tt0 < 3; ++tt0) {
      int tt = wave * 3 + tt0;
      int Mt = tt >> 3, Nt = tt & 7;
      v8f acc = {};
#pragma unroll
      for (int ks = 0; ks < 4; ++ks) {
        v16h af = frag_a_ld(sx, 128, Mt * 16, ks * 32, lane);
        v16h bf = frag_b_ld(sh_w, 128, ks * 32, Nt * 16, lane);
        acc = wmma_f16(af, bf, acc);
      }
      int gg = lane >> 4, n = Nt * 16 + (lane & 15);
      float b1 = node_b1[i * 128 + n];
#pragma unroll
      for (int v = 0; v < 8; ++v)
        sh[(Mt * 16 + 8 * gg + v) * 128 + n] = tanhf(acc[v] + b1);
    }
    __syncthreads();
    for (int t = tid; t < 48 * NT_; t += 256) {
      int row = t / NT_, tt = t % NT_;
      float acc = node_b2[i * NT_ + tt];
      const float* w2 = node_w2 + (i * NT_ + tt) * 128;
      for (int d = 0; d < 128; ++d) acc += sh[row * 128 + d] * w2[d];
      slg[row * 12 + tt] = acc;
    }
    __syncthreads();
    if (tid < 48) {
      float best = slg[tid * 12]; int arg = 0;
      for (int j = 1; j < NT_; ++j) {
        float v = slg[tid * 12 + j];
        if (v > best) { best = v; arg = j; }
      }
      sks[tid] += arg;
    }
    __syncthreads();
  }
  for (int t = tid; t < 48 * NT_; t += 256) {
    int row = t / NT_, j = t % NT_;
    int gr = base + row;
    if (gr < B_ * N_) {
      int K = sks[row] % NT_;
      int src = j - K; if (src < 0) src += NT_;
      out[gr * NT_ + j] = x_deq[gr * NT_ + src];
    }
  }
}

// ---------------- kernel 4: edge flow (12 layers -> shift of adj_deq) ------
__global__ __launch_bounds__(256) void k_edge_flow(
    const float* __restrict__ graph_sum, const float* __restrict__ selA,
    const float* __restrict__ selB, const float* __restrict__ scale,
    const float* __restrict__ shift, const _Float16* __restrict__ ew1_16,
    const float* __restrict__ edge_b1, const float* __restrict__ edge_w2,
    const float* __restrict__ edge_b2, const float* __restrict__ adj_deq,
    float* __restrict__ out) {
  __shared__ _Float16 sx[48 * 384];
  __shared__ float    sh[48 * 128];
  __shared__ float    slg[48 * 4];
  __shared__ int      sks[48];
  const int tid = threadIdx.x, wave = tid >> 5, lane = tid & 31;
  const int base = blockIdx.x * 48;

  for (int t = tid; t < 48 * 384; t += 256) {
    int row = t / 384, c = t % 384;
    int gr = base + row;
    float v = 0.0f;
    if (gr < B_ * NE_) {
      int bb = gr / NE_, e = gr % NE_;
      int d = c & 127, seg = c >> 7;
      if (seg == 0)      v = selA[gr * 128 + d] * scale[d] + shift[d];
      else if (seg == 1) v = selB[gr * 128 + d] * scale[d] + shift[d];
      else v = graph_sum[(bb * R_ + N_ + e) * 128 + d] * scale[d] +
               (float)N_ * shift[d];
    }
    sx[t] = (_Float16)v;
  }
  if (tid < 48) sks[tid] = 0;
  __syncthreads();

  for (int i = 0; i < L_; ++i) {
    const _Float16* W = ew1_16 + i * 128 * 384;
    // prefetch next layer's weights into cache (global_prefetch_b8)
    {
      const _Float16* Wn = ew1_16 + ((i + 1) % L_) * 128 * 384;
#pragma unroll
      for (int p = 0; p < 3; ++p)
        __builtin_prefetch(Wn + (tid * 3 + p) * 64, 0, 1);
    }
    for (int tt0 = 0; tt0 < 3; ++tt0) {
      int tt = wave * 3 + tt0;
      int Mt = tt >> 3, Nt = tt & 7;
      v8f acc = {};
#pragma unroll
      for (int ks = 0; ks < 12; ++ks) {
        v16h af = frag_a_ld(sx, 384, Mt * 16, ks * 32, lane);
        v16h bf = frag_b_ld(W, 384, ks * 32, Nt * 16, lane);
        acc = wmma_f16(af, bf, acc);
      }
      int gg = lane >> 4, n = Nt * 16 + (lane & 15);
      float b1 = edge_b1[i * 128 + n];
#pragma unroll
      for (int v = 0; v < 8; ++v)
        sh[(Mt * 16 + 8 * gg + v) * 128 + n] = tanhf(acc[v] + b1);
    }
    __syncthreads();
    for (int t = tid; t < 48 * ET_; t += 256) {
      int row = t / ET_, tt = t % ET_;
      float acc = edge_b2[i * ET_ + tt];
      const float* w2 = edge_w2 + (i * ET_ + tt) * 128;
      for (int d = 0; d < 128; ++d) acc += sh[row * 128 + d] * w2[d];
      slg[row * 4 + tt] = acc;
    }
    __syncthreads();
    if (tid < 48) {
      float best = slg[tid * 4]; int arg = 0;
      for (int j = 1; j < ET_; ++j) {
        float v = slg[tid * 4 + j];
        if (v > best) { best = v; arg = j; }
      }
      sks[tid] += arg;
    }
    __syncthreads();
  }
  for (int t = tid; t < 48 * ET_; t += 256) {
    int row = t / ET_, j = t % ET_;
    int gr = base + row;
    if (gr < B_ * NE_) {
      int K = sks[row] % ET_;
      int src = j - K; if (src < 0) src += ET_;
      out[gr * ET_ + j] = adj_deq[gr * ET_ + src];
    }
  }
}

// ---------------- host launcher -------------------------------------------
extern "C" void kernel_launch(void* const* d_in, const int* in_sizes, int n_in,
                              void* d_out, int out_size, void* d_ws, size_t ws_size,
                              hipStream_t stream) {
  const float* x        = (const float*)d_in[0];
  const float* adj      = (const float*)d_in[1];
  const float* x_deq    = (const float*)d_in[2];
  const float* adj_deq  = (const float*)d_in[3];
  const unsigned char* mask_node = (const unsigned char*)d_in[4];
  const unsigned char* mask_edge = (const unsigned char*)d_in[5];
  const int*   isel     = (const int*)d_in[6];
  const float* emb_w    = (const float*)d_in[7];
  const float* gc1_w    = (const float*)d_in[8];
  const float* gc2_w    = (const float*)d_in[9];
  const float* gc3_w    = (const float*)d_in[10];
  const float* bn_gamma = (const float*)d_in[11];
  const float* bn_beta  = (const float*)d_in[12];
  const float* node_w1  = (const float*)d_in[13];
  const float* node_b1  = (const float*)d_in[14];
  const float* node_w2  = (const float*)d_in[15];
  const float* node_b2  = (const float*)d_in[16];
  const float* edge_w1  = (const float*)d_in[17];
  const float* edge_b1  = (const float*)d_in[18];
  const float* edge_w2  = (const float*)d_in[19];
  const float* edge_b2  = (const float*)d_in[20];

  char* ws = (char*)d_ws;
  _Float16* gc2_16 = (_Float16*)(ws);
  _Float16* gc3_16 = (_Float16*)(ws + 98304);
  _Float16* ew1_16 = (_Float16*)(ws + 196608);
  _Float16* nw1_16 = (_Float16*)(ws + 1376256);
  float* bn_sum    = (float*)(ws + 1769472);
  float* bn_sumsq  = bn_sum + 128;
  float* scale     = bn_sum + 256;
  float* shift     = bn_sum + 384;
  float* graph_sum = (float*)(ws + 1771520);             // B*R*128 f32
  float* selA      = (float*)(ws + 1771520 + 6070272);   // B*NE*128 f32
  float* selB      = (float*)(ws + 1771520 + 6070272 + 5758976);

  float* out = (float*)d_out;

  k_prep<<<1024, 256, 0, stream>>>(gc2_w, gc3_w, edge_w1, node_w1,
                                   gc2_16, gc3_16, ew1_16, nw1_16, bn_sum);

  const size_t smem1 = 18432 + 49152 + 12288 + 1824;  // 81696 B dynamic LDS
  k_rgcn<<<B_ * R_, 256, smem1, stream>>>(
      x, adj, mask_node, mask_edge, isel, emb_w, gc1_w, gc2_16, gc3_16,
      bn_sum, bn_sumsq, graph_sum, selA, selB);

  k_bn<<<1, 128, 0, stream>>>(bn_sum, bn_sumsq, bn_gamma, bn_beta, scale, shift);

  const size_t smem3 = 72128;  // sx + sh_w + sh + slg + sks
  k_node_flow<<<(B_ * N_ + 47) / 48, 256, smem3, stream>>>(
      graph_sum, scale, shift, nw1_16, node_b1, node_w2, node_b2, x_deq, out);

  k_edge_flow<<<(B_ * NE_ + 47) / 48, 256, 0, stream>>>(
      graph_sum, selA, selB, scale, shift, ew1_16, edge_b1, edge_w2, edge_b2,
      adj_deq, out + B_ * N_ * NT_);
}
